// Attention_48103633715481
// MI455X (gfx1250) — matrix operand order repaired and checked by reading, not compiled, not run on hardware
//
#include <hip/hip_runtime.h>
#include <hip/hip_bf16.h>

typedef __attribute__((ext_vector_type(16))) __bf16 v16bf;
typedef __attribute__((ext_vector_type(8)))  float  v8f;
typedef unsigned short u16;

constexpr int kSeq   = 2048;
constexpr int kD     = 768;
constexpr int kH     = 12;
constexpr int kHd    = 64;

union BFB {
    v16bf  v;
    __bf16 e[16];
    uint4  q[2];
};

__device__ __forceinline__ u16 f2bf(float f) {
    __bf16 h = (__bf16)f;
    return __builtin_bit_cast(u16, h);
}

__device__ __forceinline__ void cvt8(__bf16* d, float4 a, float4 b) {
    d[0] = (__bf16)a.x; d[1] = (__bf16)a.y; d[2] = (__bf16)a.z; d[3] = (__bf16)a.w;
    d[4] = (__bf16)b.x; d[5] = (__bf16)b.y; d[6] = (__bf16)b.z; d[7] = (__bf16)b.w;
}

__device__ __forceinline__ v8f wmma_bf16(v16bf a, v16bf b, v8f c) {
    return __builtin_amdgcn_wmma_f32_16x16x32_bf16(
        /*neg_a=*/false, a, /*neg_b=*/false, b,
        /*c_mod=*/(short)0, c, /*reuse_a=*/false, /*reuse_b=*/false);
}

// ---------------------------------------------------------------------------
// Kernel 1: QKV projections.  C[m,n] = sum_k X[m,k] * W[n,k]
// Each wave computes one 16(M) x 64(N) tile.  blockIdx.z selects Q/K/V.
// q,k written head-major [b,h,s,d]; v written transposed [b,h,d,s].
// ---------------------------------------------------------------------------
__global__ __launch_bounds__(256) void proj_qkv_kernel(
    const float* __restrict__ Q, const float* __restrict__ K, const float* __restrict__ V,
    const float* __restrict__ Wq, const float* __restrict__ Wk, const float* __restrict__ Wv,
    u16* __restrict__ qh, u16* __restrict__ kh, u16* __restrict__ vT)
{
    const int w    = threadIdx.x >> 5;
    const int lane = threadIdx.x & 31;
    const int hl   = lane >> 4;
    const int l16  = lane & 15;
    const int which = blockIdx.z;

    const float* X = (which == 0) ? Q : (which == 1) ? K : V;
    const float* W = (which == 0) ? Wq : (which == 1) ? Wk : Wv;

    const int mBase = (blockIdx.y * 8 + w) * 16;
    const int nBase = blockIdx.x * 64;

    v8f acc[4] = {};

    for (int kc = 0; kc < kD; kc += 32) {
        // A tile: 16x32, lane = M row, two 8-elem K spans (hl*8, hl*8+16)
        const float* arow = X + (size_t)(mBase + l16) * kD + kc + hl * 8;
        BFB a;
        cvt8(a.e,     ((const float4*)arow)[0], ((const float4*)(arow + 4))[0]);
        cvt8(a.e + 8, ((const float4*)(arow + 16))[0], ((const float4*)(arow + 20))[0]);

#pragma unroll
        for (int nt = 0; nt < 4; ++nt) {
            // B tile: 32x16, lane = N col, two 8-elem K spans (hl*8, hl*8+16)
            const float* brow = W + (size_t)(nBase + nt * 16 + l16) * kD + kc + hl * 8;
            BFB bb;
            cvt8(bb.e,     ((const float4*)brow)[0], ((const float4*)(brow + 4))[0]);
            cvt8(bb.e + 8, ((const float4*)(brow + 16))[0], ((const float4*)(brow + 20))[0]);
            acc[nt] = wmma_bf16(a.v, bb.v, acc[nt]);
        }
    }

    // Scatter: C layout -> lane comp r has (M = r + 8*hl, N = l16)
#pragma unroll
    for (int nt = 0; nt < 4; ++nt) {
#pragma unroll
        for (int r = 0; r < 8; ++r) {
            const int m = mBase + r + 8 * hl;
            const int n = nBase + nt * 16 + l16;
            const int b = m >> 11, s = m & (kSeq - 1);
            const int h = n >> 6,  d = n & (kHd - 1);
            const u16 bits = f2bf(acc[nt][r]);
            if (which == 2) {
                vT[(((size_t)b * kH + h) * kHd + d) * kSeq + s] = bits;
            } else {
                u16* dst = (which == 0) ? qh : kh;
                dst[(((size_t)b * kH + h) * kSeq + s) * kHd + d] = bits;
            }
        }
    }
}

// ---------------------------------------------------------------------------
// Kernel 2: flash attention.  One wave owns a 16-query tile; loops over keys
// in blocks of 32 (two 16x16 S tiles -> one 16x32 P A-tile), online softmax,
// accumulates O (16 x 64) in 4 WMMA accumulators.  P relaid via 1KB LDS.
// ---------------------------------------------------------------------------
__global__ __launch_bounds__(128) void flash_attn_kernel(
    const u16* __restrict__ qh, const u16* __restrict__ kh, const u16* __restrict__ vT,
    const int* __restrict__ mask, u16* __restrict__ ao)
{
    __shared__ u16 pbuf[4 * 16 * 32];   // 1KB per wave

    const int w    = threadIdx.x >> 5;
    const int lane = threadIdx.x & 31;
    const int hl   = lane >> 4;
    const int l16  = lane & 15;
    const int b    = blockIdx.z;
    const int h    = blockIdx.y;
    const int qbase = (blockIdx.x * 4 + w) * 16;

    const float scale   = 0.0360843918f;           // 1/sqrt(768)
    const float maskedv = -1e20f * 0.0360843918f;  // matches reference masking

    const size_t headoff = ((size_t)b * kH + h) * (size_t)kSeq * kHd;

    // Q A-tiles (d 0..31 and 32..63), resident for the whole key loop
    BFB aq0, aq1;
    {
        const u16* qrow = qh + headoff + (size_t)(qbase + l16) * kHd;
        aq0.q[0] = *(const uint4*)(qrow + hl * 8);
        aq0.q[1] = *(const uint4*)(qrow + hl * 8 + 16);
        aq1.q[0] = *(const uint4*)(qrow + 32 + hl * 8);
        aq1.q[1] = *(const uint4*)(qrow + 32 + hl * 8 + 16);
    }

    float mrow[8], lrow[8];
#pragma unroll
    for (int r = 0; r < 8; ++r) { mrow[r] = -3.0e38f; lrow[r] = 0.f; }
    v8f o[4] = {};
    v8f zero = {};
    u16* myp = pbuf + w * 512;

    const u16* vbase = vT + headoff;   // [d][s] layout, same head offset

    for (int jb = 0; jb < kSeq; jb += 32) {
        // two 16x16 S tiles (keys jb..jb+15, jb+16..jb+31)
        v8f s[2];
#pragma unroll
        for (int t = 0; t < 2; ++t) {
            const u16* krow = kh + headoff + (size_t)(jb + t * 16 + l16) * kHd;
            BFB bk0, bk1;
            bk0.q[0] = *(const uint4*)(krow + hl * 8);
            bk0.q[1] = *(const uint4*)(krow + hl * 8 + 16);
            bk1.q[0] = *(const uint4*)(krow + 32 + hl * 8);
            bk1.q[1] = *(const uint4*)(krow + 32 + hl * 8 + 16);
            s[t] = wmma_bf16(aq0.v, bk0.v, zero);
            s[t] = wmma_bf16(aq1.v, bk1.v, s[t]);
        }

        // mask + scale (reference: mask -> -1e20, then * scale)
#pragma unroll
        for (int r = 0; r < 8; ++r) {
            const int qi = qbase + r + 8 * hl;
            const int* mp = mask + (size_t)qi * kSeq + jb + l16;
            s[0][r] = mp[0]  ? s[0][r] * scale : maskedv;
            s[1][r] = mp[16] ? s[1][r] * scale : maskedv;
        }

        // online softmax: rows live across the 16 lanes of each half-wave
#pragma unroll
        for (int r = 0; r < 8; ++r) {
            float mt = fmaxf(s[0][r], s[1][r]);
#pragma unroll
            for (int dd = 1; dd < 16; dd <<= 1) mt = fmaxf(mt, __shfl_xor(mt, dd, 16));
            const float mnew  = fmaxf(mrow[r], mt);
            const float alpha = __expf(mrow[r] - mnew);
            const float p0 = __expf(s[0][r] - mnew);
            const float p1 = __expf(s[1][r] - mnew);
            s[0][r] = p0; s[1][r] = p1;
            float rs = p0 + p1;
#pragma unroll
            for (int dd = 1; dd < 16; dd <<= 1) rs += __shfl_xor(rs, dd, 16);
            lrow[r] = lrow[r] * alpha + rs;
            mrow[r] = mnew;
            o[0][r] *= alpha; o[1][r] *= alpha; o[2][r] *= alpha; o[3][r] *= alpha;
        }

        // P (C-layout) -> LDS -> A-layout bf16
        __syncthreads();
#pragma unroll
        for (int r = 0; r < 8; ++r) {
            const int row = r + 8 * hl;
            myp[row * 32 + l16]      = f2bf(s[0][r]);
            myp[row * 32 + 16 + l16] = f2bf(s[1][r]);
        }
        __syncthreads();
        BFB ap;
        ap.q[0] = *(const uint4*)(myp + l16 * 32 + hl * 8);
        ap.q[1] = *(const uint4*)(myp + l16 * 32 + hl * 8 + 16);

        // O += P * V  (B from transposed V: two 16B spans per lane at hl*8, hl*8+16)
#pragma unroll
        for (int t = 0; t < 4; ++t) {
            const u16* vrow = vbase + (size_t)(t * 16 + l16) * kSeq + jb + hl * 8;
            BFB bv;
            bv.q[0] = *(const uint4*)(vrow);
            bv.q[1] = *(const uint4*)(vrow + 16);
            o[t] = wmma_bf16(ap.v, bv.v, o[t]);
        }
    }

    // normalize and write attention output, token-major [b*S+s, h*64+d]
#pragma unroll
    for (int r = 0; r < 8; ++r) {
        const float inv = 1.0f / lrow[r];
        const int token = b * kSeq + qbase + r + 8 * hl;
        u16* orow = ao + (size_t)token * kD + h * kHd + l16;
        orow[0]  = f2bf(o[0][r] * inv);
        orow[16] = f2bf(o[1][r] * inv);
        orow[32] = f2bf(o[2][r] * inv);
        orow[48] = f2bf(o[3][r] * inv);
    }
}

// ---------------------------------------------------------------------------
// Kernel 3: output projection.  out[m,n] = sum_k AO[m,k] * Wo[n,k], fp32 out.
// ---------------------------------------------------------------------------
__global__ __launch_bounds__(256) void out_proj_kernel(
    const u16* __restrict__ ao, const float* __restrict__ Wo, float* __restrict__ out)
{
    const int w    = threadIdx.x >> 5;
    const int lane = threadIdx.x & 31;
    const int hl   = lane >> 4;
    const int l16  = lane & 15;
    const int mBase = (blockIdx.y * 8 + w) * 16;
    const int nBase = blockIdx.x * 64;

    v8f acc[4] = {};

    for (int kc = 0; kc < kD; kc += 32) {
        const u16* arow = ao + (size_t)(mBase + l16) * kD + kc + hl * 8;
        BFB a;
        a.q[0] = *(const uint4*)(arow);
        a.q[1] = *(const uint4*)(arow + 16);

#pragma unroll
        for (int nt = 0; nt < 4; ++nt) {
            const float* brow = Wo + (size_t)(nBase + nt * 16 + l16) * kD + kc + hl * 8;
            BFB bb;
            cvt8(bb.e,     ((const float4*)brow)[0], ((const float4*)(brow + 4))[0]);
            cvt8(bb.e + 8, ((const float4*)(brow + 16))[0], ((const float4*)(brow + 20))[0]);
            acc[nt] = wmma_bf16(a.v, bb.v, acc[nt]);
        }
    }

#pragma unroll
    for (int nt = 0; nt < 4; ++nt)
#pragma unroll
        for (int r = 0; r < 8; ++r)
            out[(size_t)(mBase + r + 8 * hl) * kD + nBase + nt * 16 + l16] = acc[nt][r];
}

// ---------------------------------------------------------------------------
extern "C" void kernel_launch(void* const* d_in, const int* in_sizes, int n_in,
                              void* d_out, int out_size, void* d_ws, size_t ws_size,
                              hipStream_t stream) {
    const float* Q    = (const float*)d_in[0];
    const float* K    = (const float*)d_in[1];
    const float* V    = (const float*)d_in[2];
    const int*   mask = (const int*)  d_in[3];
    const float* Wq   = (const float*)d_in[4];
    const float* Wk   = (const float*)d_in[5];
    const float* Wv   = (const float*)d_in[6];
    const float* Wo   = (const float*)d_in[7];
    float* out = (float*)d_out;

    const size_t headElems = (size_t)2 * kH * kSeq * kHd;  // 3,145,728
    u16* qh = (u16*)d_ws;
    u16* kh = qh + headElems;
    u16* vT = kh + headElems;
    u16* ao = vT + headElems;

    // 1) QKV projections: grid (N-tiles=12, M-blocks=32, qkv=3), 8 waves/block
    proj_qkv_kernel<<<dim3(12, 32, 3), 256, 0, stream>>>(Q, K, V, Wq, Wk, Wv, qh, kh, vT);
    // 2) flash attention: grid (q-blocks of 64 rows = 32, heads=12, batch=2)
    flash_attn_kernel<<<dim3(kSeq / 64, kH, 2), 128, 0, stream>>>(qh, kh, vT, mask, ao);
    // 3) output projection
    out_proj_kernel<<<dim3(12, 32), 256, 0, stream>>>(ao, Wo, out);
}
